// PrimsSolver_59304908423635
// MI455X (gfx1250) — compile-verified, matrix operand order checked by reading, not run on hardware
//
#include <hip/hip_runtime.h>
#include <hip/hip_bf16.h>
#include <math.h>

typedef __attribute__((ext_vector_type(16))) _Float16 v16h;
typedef __attribute__((ext_vector_type(8)))  float    v8f;

#define NEGV   (-1000000000.0f)
#define L      32
#define NNODE  32
#define NGRAPH 256
#define NTOT   8192
#define STEPS  32

// Order-preserving float<->uint key so segment_max can use ds_max_u32 LDS atomics.
static __device__ __forceinline__ unsigned f2key(float f) {
  unsigned u = __float_as_uint(f);
  return (u & 0x80000000u) ? ~u : (u | 0x80000000u);
}
static __device__ __forceinline__ float key2f(unsigned k) {
  return __uint_as_float((k & 0x80000000u) ? (k & 0x7fffffffu) : ~k);
}
static __device__ __forceinline__ float lrelu(float v) { return v > 0.f ? v : 0.01f * v; }
static __device__ __forceinline__ float sigm(float v)  { return 1.f / (1.f + __expf(-v)); }

// Build a B operand (32x16 K x N slice, f16) from a row-major f32 weight matrix:
// B[k][n] = W[cBase + n][kBase + k].  Per-lane: n = lane&15, K pattern follows the
// documented 16-bit operand layout (K 0..7,16..23 for lanes<16; +8 for lanes>=16).
static __device__ __forceinline__ v16h makeB(const float* W, int stride, int cBase,
                                             int kBase, int lane) {
  v16h b;
  const int c  = cBase + (lane & 15);
  const int k0 = kBase + ((lane >= 16) ? 8 : 0);
#pragma unroll
  for (int h = 0; h < 8; ++h) b[h]     = (_Float16)W[c * stride + k0 + h];
#pragma unroll
  for (int h = 0; h < 8; ++h) b[8 + h] = (_Float16)W[c * stride + k0 + 16 + h];
  return b;
}

// Build an A operand row (16x32, f16) from an LDS row of 32 halves (M = lane&15).
static __device__ __forceinline__ v16h makeA(const _Float16* row, int lane) {
  const int k0 = (lane >= 16) ? 8 : 0;
  v16h a;
#pragma unroll
  for (int h = 0; h < 8; ++h) a[h]     = row[k0 + h];
#pragma unroll
  for (int h = 0; h < 8; ++h) a[8 + h] = row[16 + k0 + h];
  return a;
}

static __device__ __forceinline__ v8f wmma_f16(v16h a, v16h b, v8f c) {
  return __builtin_amdgcn_wmma_f32_16x16x32_f16(false, a, false, b, (short)0, c,
                                                false, false);
}

__global__ void fill_neg_kernel(float4* __restrict__ out, int n4) {
  int idx = blockIdx.x * blockDim.x + threadIdx.x;
  if (idx < n4) out[idx] = make_float4(NEGV, NEGV, NEGV, NEGV);
}

__global__ __launch_bounds__(256) void prims_solver_kernel(
    const float* __restrict__ x,    const float* __restrict__ ea_g,
    const float* __restrict__ Wenc, const float* __restrict__ Wm1,
    const float* __restrict__ Wm2,  const float* __restrict__ Wu,
    const float* __restrict__ Wih,  const float* __restrict__ Whh,
    const float* __restrict__ Wmst, const float* __restrict__ Wp1,
    const float* __restrict__ Wp2,  float* __restrict__ out) {
  __shared__ float     sh_h[NNODE * L];
  __shared__ float     sh_hn[NNODE * L];
  __shared__ float     sh_enc[NNODE * L];
  __shared__ _Float16  sh_ench[NNODE * L];
  __shared__ float     sh_u[NNODE * L];
  __shared__ float     sh_prev[NNODE];
  __shared__ float     sh_ea[NNODE * NNODE];
  __shared__ unsigned  sh_aggr[NNODE * L];   // keys, then decoded f32 bits in place
  __shared__ float     sh_sc[NNODE];
  __shared__ _Float16  sh_stag[8][16 * 32];  // per-wave layer1->layer2 staging
  __shared__ float     sh_red[8][16 * 16];   // per-wave final dot reduction

  const int g    = blockIdx.x;
  const int tid  = threadIdx.x;
  const int lane = tid & 31;
  const int wv   = tid >> 5;

  // ---- init state ----
#pragma unroll
  for (int i = 0; i < 4; ++i) sh_h[tid + 256 * i] = 0.f;
  if (tid < NNODE) sh_prev[tid] = x[(g * NNODE + tid) * STEPS + 0];
#pragma unroll
  for (int i = 0; i < 4; ++i) {
    int e = tid + 256 * i;
    sh_ea[e] = ea_g[g * 1024 + e];
  }

  // Loop-invariant WMMA B operands (held in registers for all 32 steps).
  const v16h B_m1_d0 = makeB(Wm1, 65, 0, 0, lane);    // enc[dst] chunk, cols 0..15
  const v16h B_m1_d1 = makeB(Wm1, 65, 16, 0, lane);   // enc[dst] chunk, cols 16..31
  const v16h B_m1_s0 = makeB(Wm1, 65, 0, 32, lane);   // enc[src] chunk, cols 0..15
  const v16h B_m1_s1 = makeB(Wm1, 65, 16, 32, lane);  // enc[src] chunk, cols 16..31
  const v16h B_m2_0  = makeB(Wm2, 32, 0, 0, lane);
  const v16h B_m2_1  = makeB(Wm2, 32, 16, 0, lane);
  const float wc64a  = Wm1[(lane & 15) * 65 + 64];        // ea rank-1 column
  const float wc64b  = Wm1[((lane & 15) + 16) * 65 + 64];
  const unsigned KEYNEG = f2key(-1.0e30f);

  __syncthreads();

  for (int step = 0; step < STEPS; ++step) {
    // ---- enc = relu([prev, h] @ Wenc^T); also reset aggr keys ----
#pragma unroll
    for (int i = 0; i < 4; ++i) {
      int p = tid + 256 * i, nd = p >> 5, c = p & 31;
      float acc = sh_prev[nd] * Wenc[c * 33];
      for (int k = 0; k < L; ++k) acc += sh_h[nd * L + k] * Wenc[c * 33 + 1 + k];
      acc = acc > 0.f ? acc : 0.f;
      sh_enc[p]  = acc;
      sh_ench[p] = (_Float16)acc;
      sh_aggr[p] = KEYNEG;
    }
    __syncthreads();

    // ---- edge MLP (WMMA f16) + segment_max via ds_max_u32 ----
    // Tile t: 16 edges, fixed src s = t>>1, dst = j0..j0+15.
    for (int tt = 0; tt < 8; ++tt) {
      const int t = wv * 8 + tt;
      const int s = t >> 1, j0 = (t & 1) << 4;
      const v16h A_dst = makeA(&sh_ench[(j0 + (lane & 15)) * L], lane);
      const v16h A_src = makeA(&sh_ench[s * L], lane);  // replicated row
      v8f acc0, acc1;
#pragma unroll
      for (int r = 0; r < 8; ++r) {  // C init = ea ⊗ Wm1[:,64]
        int   m   = r + ((lane >> 4) << 3);
        float eav = sh_ea[t * 16 + m];
        acc0[r] = eav * wc64a;
        acc1[r] = eav * wc64b;
      }
      acc0 = wmma_f16(A_dst, B_m1_d0, acc0);
      acc0 = wmma_f16(A_src, B_m1_s0, acc0);
      acc1 = wmma_f16(A_dst, B_m1_d1, acc1);
      acc1 = wmma_f16(A_src, B_m1_s1, acc1);

      _Float16* stg = sh_stag[wv];  // C-layout -> A-layout reshuffle via LDS
#pragma unroll
      for (int r = 0; r < 8; ++r) {
        int m = r + ((lane >> 4) << 3);
        stg[m * 32 + (lane & 15)]      = (_Float16)lrelu(acc0[r]);
        stg[m * 32 + 16 + (lane & 15)] = (_Float16)lrelu(acc1[r]);
      }
      asm volatile("s_wait_dscnt 0" ::: "memory");

      const v16h A_hid = makeA(&stg[(lane & 15) * 32], lane);
      v8f zc = {};
      v8f m0 = wmma_f16(A_hid, B_m2_0, zc);
      v8f m1 = wmma_f16(A_hid, B_m2_1, zc);
#pragma unroll
      for (int r = 0; r < 8; ++r) {
        int m = r + ((lane >> 4) << 3);
        int dst = j0 + m;
        atomicMax(&sh_aggr[dst * L + (lane & 15)],      f2key(lrelu(m0[r])));
        atomicMax(&sh_aggr[dst * L + 16 + (lane & 15)], f2key(lrelu(m1[r])));
      }
    }
    __syncthreads();

    // ---- decode aggr keys to floats in place ----
#pragma unroll
    for (int i = 0; i < 4; ++i) {
      int p = tid + 256 * i;
      sh_aggr[p] = __float_as_uint(key2f(sh_aggr[p]));
    }
    __syncthreads();
    const float* aggr = (const float*)sh_aggr;

    // ---- u = lrelu([enc, aggr] @ Wu^T) ----
#pragma unroll
    for (int i = 0; i < 4; ++i) {
      int p = tid + 256 * i, nd = p >> 5, c = p & 31;
      float acc = 0.f;
      for (int k = 0; k < L; ++k) acc += sh_enc[nd * L + k] * Wu[c * 64 + k];
      for (int k = 0; k < L; ++k) acc += aggr[nd * L + k] * Wu[c * 64 + 32 + k];
      sh_u[p] = lrelu(acc);
    }
    __syncthreads();

    // ---- GRU ----
#pragma unroll
    for (int i = 0; i < 4; ++i) {
      int p = tid + 256 * i, nd = p >> 5, c = p & 31;
      float ir = 0, iz = 0, in_ = 0, hr = 0, hz = 0, hn = 0;
      for (int k = 0; k < L; ++k) {
        float uv = sh_u[nd * L + k], hv = sh_h[nd * L + k];
        ir  += uv * Wih[c * L + k];
        iz  += uv * Wih[(c + 32) * L + k];
        in_ += uv * Wih[(c + 64) * L + k];
        hr  += hv * Whh[c * L + k];
        hz  += hv * Whh[(c + 32) * L + k];
        hn  += hv * Whh[(c + 64) * L + k];
      }
      float r  = sigm(ir + hr), zz = sigm(iz + hz);
      float nv = tanhf(in_ + r * hn);
      sh_hn[p] = (1.f - zz) * nv + zz * sh_h[p];
    }
    __syncthreads();

    // ---- mst score, mask, argmax, prev update ----
    if (tid < NNODE) {
      float acc = 0.f;
      for (int k = 0; k < L; ++k)
        acc += sh_enc[tid * L + k] * Wmst[k] + sh_hn[tid * L + k] * Wmst[32 + k];
      sh_sc[tid] = (sh_prev[tid] > 0.f) ? NEGV : acc;
    }
    __syncthreads();
    if (tid == 0) {
      int best = 0;
      float bv = sh_sc[0];
      for (int j = 1; j < NNODE; ++j)
        if (sh_sc[j] > bv) { bv = sh_sc[j]; best = j; }
      sh_prev[best] = 1.f;
    }
    __syncthreads();
#pragma unroll
    for (int i = 0; i < 4; ++i) { int p = tid + 256 * i; sh_h[p] = sh_hn[p]; }
    __syncthreads();
  }

  // ---- final edge predictor: out[e] = relu([h[src],h[dst],ea] @ Wp1^T) @ Wp2^T ----
#pragma unroll
  for (int i = 0; i < 4; ++i) {
    int p = tid + 256 * i;
    sh_ench[p] = (_Float16)sh_h[p];
  }
  __syncthreads();

  const v16h B_p1_s0 = makeB(Wp1, 65, 0, 0, lane);    // x h[src]
  const v16h B_p1_s1 = makeB(Wp1, 65, 16, 0, lane);
  const v16h B_p1_d0 = makeB(Wp1, 65, 0, 32, lane);   // x h[dst]
  const v16h B_p1_d1 = makeB(Wp1, 65, 16, 32, lane);
  const float wpc64a = Wp1[(lane & 15) * 65 + 64];
  const float wpc64b = Wp1[((lane & 15) + 16) * 65 + 64];
  const float w2a = Wp2[lane & 15];
  const float w2b = Wp2[16 + (lane & 15)];

  for (int tt = 0; tt < 8; ++tt) {
    const int t = wv * 8 + tt;
    const int s = t >> 1, j0 = (t & 1) << 4;
    const v16h A_src = makeA(&sh_ench[s * L], lane);
    const v16h A_dst = makeA(&sh_ench[(j0 + (lane & 15)) * L], lane);
    v8f acc0, acc1;
#pragma unroll
    for (int r = 0; r < 8; ++r) {
      int   m   = r + ((lane >> 4) << 3);
      float eav = sh_ea[t * 16 + m];
      acc0[r] = eav * wpc64a;
      acc1[r] = eav * wpc64b;
    }
    acc0 = wmma_f16(A_src, B_p1_s0, acc0);
    acc0 = wmma_f16(A_dst, B_p1_d0, acc0);
    acc1 = wmma_f16(A_src, B_p1_s1, acc1);
    acc1 = wmma_f16(A_dst, B_p1_d1, acc1);

    float* red = sh_red[wv];
#pragma unroll
    for (int r = 0; r < 8; ++r) {
      int   m  = r + ((lane >> 4) << 3);
      float h0 = acc0[r] > 0.f ? acc0[r] : 0.f;
      float h1 = acc1[r] > 0.f ? acc1[r] : 0.f;
      red[m * 16 + (lane & 15)] = h0 * w2a + h1 * w2b;
    }
    asm volatile("s_wait_dscnt 0" ::: "memory");
    if (lane < 16) {
      int   m = lane;
      float o = 0.f;
      for (int c = 0; c < 16; ++c) o += red[m * 16 + c];
      int row = g * NNODE + s;
      int col = g * NNODE + j0 + m;
      out[(size_t)row * NTOT + col] = o;
    }
  }
}

extern "C" void kernel_launch(void* const* d_in, const int* in_sizes, int n_in,
                              void* d_out, int out_size, void* d_ws, size_t ws_size,
                              hipStream_t stream) {
  (void)in_sizes; (void)n_in; (void)d_ws; (void)ws_size;
  const float* x    = (const float*)d_in[0];
  const float* ea   = (const float*)d_in[1];
  const float* Wenc = (const float*)d_in[2];
  const float* Wm1  = (const float*)d_in[3];
  const float* Wm2  = (const float*)d_in[4];
  const float* Wu   = (const float*)d_in[5];
  const float* Wih  = (const float*)d_in[6];
  const float* Whh  = (const float*)d_in[7];
  const float* Wmst = (const float*)d_in[8];
  const float* Wp1  = (const float*)d_in[9];
  const float* Wp2  = (const float*)d_in[10];
  float* out = (float*)d_out;

  // 256 MB NEG fill: pure HBM-bandwidth kernel (~11 us at 23.3 TB/s).
  int n4 = out_size / 4;
  fill_neg_kernel<<<(n4 + 255) / 256, 256, 0, stream>>>((float4*)out, n4);

  // One workgroup (8 wave32s) per graph; all state LDS-resident; WMMA edge MLP.
  prims_solver_kernel<<<NGRAPH, 256, 0, stream>>>(x, ea, Wenc, Wm1, Wm2, Wu, Wih,
                                                  Whh, Wmst, Wp1, Wp2, out);
}
